// SelfAttentionVanilla_63960652972222
// MI455X (gfx1250) — matrix operand order repaired and checked by reading, not compiled, not run on hardware
//
#include <hip/hip_runtime.h>

// ---------------------------------------------------------------------------
// Flash attention forward for B=4, H=16, S=2048, D=64 (fp32 in/out).
// CDNA5 / gfx1250: wave32, V_WMMA_F32_16X16X32_BF16 for both GEMMs.
// 4 waves/block, 32 Q rows per wave (two 16-row stripes) so every staged
// K/V B-fragment feeds two WMMAs; f32 online softmax in the exp2 domain with
// single-instruction DPP butterfly reductions and raw v_exp_f32.
// ---------------------------------------------------------------------------

typedef __attribute__((ext_vector_type(16))) __bf16 v16bf;
typedef __attribute__((ext_vector_type(8)))  __bf16 v8bf;
typedef __attribute__((ext_vector_type(8)))  float  v8f;

#define ATTN_B   4
#define ATTN_H   16
#define ATTN_S   2048
#define ATTN_D   64
#define KV_TILE  32
#define QSTRIPE  2                    // 16-row stripes per wave
#define Q_TILE   128                  // 4 waves * 2 stripes * 16 rows
#define NWAVES   4
#define NKVT     (ATTN_S / KV_TILE)

// Fused VOP2+DPP butterfly steps: dst = op(dpp(src), src) in ONE VALU op.
// DPP rows are 16 lanes, matching the WMMA C-fragment halves on wave32.
__device__ __forceinline__ float row16_max(float x) {
  float t;
  asm("v_max_num_f32_dpp %0, %1, %1 quad_perm:[1,0,3,2] row_mask:0xf bank_mask:0xf bound_ctrl:1"
      : "=v"(t) : "v"(x));
  asm("v_max_num_f32_dpp %0, %1, %1 quad_perm:[2,3,0,1] row_mask:0xf bank_mask:0xf bound_ctrl:1"
      : "=v"(x) : "v"(t));
  asm("v_max_num_f32_dpp %0, %1, %1 row_half_mirror row_mask:0xf bank_mask:0xf bound_ctrl:1"
      : "=v"(t) : "v"(x));
  asm("v_max_num_f32_dpp %0, %1, %1 row_mirror row_mask:0xf bank_mask:0xf bound_ctrl:1"
      : "=v"(x) : "v"(t));
  return x;
}

__device__ __forceinline__ float row16_sum(float x) {
  float t;
  asm("v_add_f32_dpp %0, %1, %1 quad_perm:[1,0,3,2] row_mask:0xf bank_mask:0xf bound_ctrl:1"
      : "=v"(t) : "v"(x));
  asm("v_add_f32_dpp %0, %1, %1 quad_perm:[2,3,0,1] row_mask:0xf bank_mask:0xf bound_ctrl:1"
      : "=v"(x) : "v"(t));
  asm("v_add_f32_dpp %0, %1, %1 row_half_mirror row_mask:0xf bank_mask:0xf bound_ctrl:1"
      : "=v"(t) : "v"(x));
  asm("v_add_f32_dpp %0, %1, %1 row_mirror row_mask:0xf bank_mask:0xf bound_ctrl:1"
      : "=v"(x) : "v"(t));
  return x;
}

// Raw base-2 hardware exponential (softmax weights don't need the OCML
// subnormal-range fixup: arguments are <= 0 and underflow to 0 is correct).
#if __has_builtin(__builtin_amdgcn_exp2f)
__device__ __forceinline__ float fast_exp2(float x) {
  return __builtin_amdgcn_exp2f(x);
}
#else
__device__ __forceinline__ float fast_exp2(float x) {
  float r;
  asm volatile("v_exp_f32 %0, %1\n\tv_nop\n\tv_nop" : "=v"(r) : "v"(x));
  return r;
}
#endif

__global__ __launch_bounds__(128) void flash_attn_fwd_gfx1250(
    const float* __restrict__ q, const float* __restrict__ k,
    const float* __restrict__ v, float* __restrict__ o) {
  // LDS: K tile row-major [kv][d], V tile transposed [d][kv] so that WMMA
  // B-fragments are two contiguous 8-bf16 runs per lane (ds_load_b128 pairs).
  __shared__ __align__(32) __bf16 Ktile[KV_TILE][ATTN_D];                 // 4 KB
  __shared__ __align__(32) __bf16 Vt[ATTN_D][KV_TILE];                    // 4 KB
  __shared__ __align__(32) __bf16 Ptile[NWAVES * QSTRIPE][16][KV_TILE];   // 8 KB

  const int tid  = threadIdx.x;
  const int wid  = tid >> 5;      // wave 0..3
  const int lane = tid & 31;
  const int half = lane >> 4;     // 0: lanes 0-15, 1: lanes 16-31
  const int ln   = lane & 15;

  const int bh    = blockIdx.y;                 // fused batch*head
  const int qblk  = blockIdx.x * Q_TILE;
  const int qrow0 = qblk + wid * (16 * QSTRIPE);  // first Q row of this wave

  // 1/sqrt(64) * log2(e): scores live in the exp2 domain so the softmax
  // exponentials map straight onto v_exp_f32 (base-2 hardware exp).
  const float qscale = 0.125f * 1.44269504088896340736f;

  const float* qb = q + (size_t)bh * ATTN_S * ATTN_D;
  const float* kb = k + (size_t)bh * ATTN_S * ATTN_D;
  const float* vb = v + (size_t)bh * ATTN_S * ATTN_D;
  float*       ob = o + (size_t)bh * ATTN_S * ATTN_D;

  // ---- Q A-fragments: per stripe, 16x32 bf16 for d 0..31 / 32..63 ----
  // A layout (16-bit, 16x32): lane half 0 holds K 0..7 and 16..23,
  // lane half 1 holds K 8..15 and 24..31, row = lane%16.
  v16bf aq[QSTRIPE][2];
#pragma unroll
  for (int s = 0; s < QSTRIPE; ++s) {
    const float* qr = qb + (size_t)(qrow0 + s * 16 + ln) * ATTN_D;
#pragma unroll
    for (int f = 0; f < 2; ++f) {
      const int c0 = f * 32 + half * 8;
      const int c1 = f * 32 + 16 + half * 8;
      const float4 x0 = reinterpret_cast<const float4*>(qr + c0)[0];
      const float4 x1 = reinterpret_cast<const float4*>(qr + c0)[1];
      const float4 y0 = reinterpret_cast<const float4*>(qr + c1)[0];
      const float4 y1 = reinterpret_cast<const float4*>(qr + c1)[1];
      v16bf raw;
      raw[0]  = (__bf16)(x0.x * qscale); raw[1]  = (__bf16)(x0.y * qscale);
      raw[2]  = (__bf16)(x0.z * qscale); raw[3]  = (__bf16)(x0.w * qscale);
      raw[4]  = (__bf16)(x1.x * qscale); raw[5]  = (__bf16)(x1.y * qscale);
      raw[6]  = (__bf16)(x1.z * qscale); raw[7]  = (__bf16)(x1.w * qscale);
      raw[8]  = (__bf16)(y0.x * qscale); raw[9]  = (__bf16)(y0.y * qscale);
      raw[10] = (__bf16)(y0.z * qscale); raw[11] = (__bf16)(y0.w * qscale);
      raw[12] = (__bf16)(y1.x * qscale); raw[13] = (__bf16)(y1.y * qscale);
      raw[14] = (__bf16)(y1.z * qscale); raw[15] = (__bf16)(y1.w * qscale);
      aq[s][f] = raw;
    }
  }

  // Online-softmax state per stripe: lane holds rows r+8*half, r=0..7,
  // matching the WMMA C-fragment row mapping.
  float m_i[QSTRIPE][8], l_i[QSTRIPE][8];
  v8f acc[QSTRIPE][4];   // per stripe: 16x64 f32 as four 16x16 fragments
#pragma unroll
  for (int s = 0; s < QSTRIPE; ++s) {
#pragma unroll
    for (int r = 0; r < 8; ++r) { m_i[s][r] = -__builtin_inff(); l_i[s][r] = 0.0f; }
#pragma unroll
    for (int d = 0; d < 4; ++d)
      acc[s][d] = (v8f){0.f, 0.f, 0.f, 0.f, 0.f, 0.f, 0.f, 0.f};
  }

  for (int kt = 0; kt < NKVT; ++kt) {
    __syncthreads();   // previous iteration's K/V reads done before restage

    // ---- cooperative staging: 128 threads, 32x64 f32 -> bf16 tiles ----
    {
      const int row  = tid >> 2;            // 0..31 (kv row in tile)
      const int dseg = (tid & 3) * 16;      // 0,16,32,48
      const float* kr = kb + (size_t)(kt * KV_TILE + row) * ATTN_D + dseg;
      const float* vr = vb + (size_t)(kt * KV_TILE + row) * ATTN_D + dseg;

      const float4 k0 = reinterpret_cast<const float4*>(kr)[0];
      const float4 k1 = reinterpret_cast<const float4*>(kr)[1];
      const float4 k2 = reinterpret_cast<const float4*>(kr)[2];
      const float4 k3 = reinterpret_cast<const float4*>(kr)[3];
      v8bf p0 = {(__bf16)k0.x, (__bf16)k0.y, (__bf16)k0.z, (__bf16)k0.w,
                 (__bf16)k1.x, (__bf16)k1.y, (__bf16)k1.z, (__bf16)k1.w};
      v8bf p1 = {(__bf16)k2.x, (__bf16)k2.y, (__bf16)k2.z, (__bf16)k2.w,
                 (__bf16)k3.x, (__bf16)k3.y, (__bf16)k3.z, (__bf16)k3.w};
      *reinterpret_cast<v8bf*>(&Ktile[row][dseg + 0]) = p0;
      *reinterpret_cast<v8bf*>(&Ktile[row][dseg + 8]) = p1;

      const float4 v0 = reinterpret_cast<const float4*>(vr)[0];
      const float4 v1 = reinterpret_cast<const float4*>(vr)[1];
      const float4 v2 = reinterpret_cast<const float4*>(vr)[2];
      const float4 v3 = reinterpret_cast<const float4*>(vr)[3];
      Vt[dseg +  0][row] = (__bf16)v0.x; Vt[dseg +  1][row] = (__bf16)v0.y;
      Vt[dseg +  2][row] = (__bf16)v0.z; Vt[dseg +  3][row] = (__bf16)v0.w;
      Vt[dseg +  4][row] = (__bf16)v1.x; Vt[dseg +  5][row] = (__bf16)v1.y;
      Vt[dseg +  6][row] = (__bf16)v1.z; Vt[dseg +  7][row] = (__bf16)v1.w;
      Vt[dseg +  8][row] = (__bf16)v2.x; Vt[dseg +  9][row] = (__bf16)v2.y;
      Vt[dseg + 10][row] = (__bf16)v2.z; Vt[dseg + 11][row] = (__bf16)v2.w;
      Vt[dseg + 12][row] = (__bf16)v3.x; Vt[dseg + 13][row] = (__bf16)v3.y;
      Vt[dseg + 14][row] = (__bf16)v3.z; Vt[dseg + 15][row] = (__bf16)v3.w;

      if (kt + 1 < NKVT) {   // gfx1250 global_prefetch_b8 of next tile
        __builtin_prefetch(kr + KV_TILE * ATTN_D, 0, 1);
        __builtin_prefetch(vr + KV_TILE * ATTN_D, 0, 1);
      }
    }
    __syncthreads();

    // ---- S = (Q*qscale) K^T : per stripe two 16x16 f32 fragments.
    // B layout (16-bit, 32x16): lane col = lane%16, K rows half*8..+7 and 16+half*8..+7.
    // Each K B-fragment is loaded once and feeds both Q stripes.
    v8f sc[QSTRIPE][2];
#pragma unroll
    for (int nb = 0; nb < 2; ++nb) {
      const v16bf bk0 = __builtin_shufflevector(
          *reinterpret_cast<const v8bf*>(&Ktile[nb * 16 + ln][0 * 32 + half * 8]),
          *reinterpret_cast<const v8bf*>(&Ktile[nb * 16 + ln][0 * 32 + 16 + half * 8]),
          0, 1, 2, 3, 4, 5, 6, 7, 8, 9, 10, 11, 12, 13, 14, 15);
      const v16bf bk1 = __builtin_shufflevector(
          *reinterpret_cast<const v8bf*>(&Ktile[nb * 16 + ln][1 * 32 + half * 8]),
          *reinterpret_cast<const v8bf*>(&Ktile[nb * 16 + ln][1 * 32 + 16 + half * 8]),
          0, 1, 2, 3, 4, 5, 6, 7, 8, 9, 10, 11, 12, 13, 14, 15);
#pragma unroll
      for (int s = 0; s < QSTRIPE; ++s) {
        v8f t = (v8f){0.f, 0.f, 0.f, 0.f, 0.f, 0.f, 0.f, 0.f};
        t = __builtin_amdgcn_wmma_f32_16x16x32_bf16(false, aq[s][0], false, bk0,
                                                    (short)0, t, false, false);
        t = __builtin_amdgcn_wmma_f32_16x16x32_bf16(false, aq[s][1], false, bk1,
                                                    (short)0, t, false, false);
        sc[s][nb] = t;
      }
    }

    // ---- online softmax (exp2 domain): one-instruction DPP butterfly
    // reductions across the 16-lane row holding each score row ----
#pragma unroll
    for (int s = 0; s < QSTRIPE; ++s) {
#pragma unroll
      for (int r = 0; r < 8; ++r) {
        const float mx    = row16_max(fmaxf(sc[s][0][r], sc[s][1][r]));
        const float m_new = fmaxf(m_i[s][r], mx);
        const float alpha = fast_exp2(m_i[s][r] - m_new);
        const float e0    = fast_exp2(sc[s][0][r] - m_new);
        const float e1    = fast_exp2(sc[s][1][r] - m_new);
        const float rs    = row16_sum(e0 + e1);
        l_i[s][r] = l_i[s][r] * alpha + rs;
        m_i[s][r] = m_new;
#pragma unroll
        for (int d = 0; d < 4; ++d) acc[s][d][r] *= alpha;
        // stage P (bf16) row-major into per-wave LDS for C->A relayout;
        // same-wave LDS ops are in-order, no barrier needed.
        const int prow = r + 8 * half;
        Ptile[wid * QSTRIPE + s][prow][ln]      = (__bf16)e0;
        Ptile[wid * QSTRIPE + s][prow][16 + ln] = (__bf16)e1;
      }
    }

    // ---- O += P V : P as A-fragment (16x32), Vt rows as B-fragments.
    // Each V B-fragment is loaded once and feeds both Q stripes.
    v16bf pa[QSTRIPE];
#pragma unroll
    for (int s = 0; s < QSTRIPE; ++s) {
      const v8bf plo = *reinterpret_cast<const v8bf*>(
          &Ptile[wid * QSTRIPE + s][ln][half * 8]);
      const v8bf phi = *reinterpret_cast<const v8bf*>(
          &Ptile[wid * QSTRIPE + s][ln][16 + half * 8]);
      pa[s] = __builtin_shufflevector(
          plo, phi, 0, 1, 2, 3, 4, 5, 6, 7, 8, 9, 10, 11, 12, 13, 14, 15);
    }
#pragma unroll
    for (int db = 0; db < 4; ++db) {
      const v16bf bv = __builtin_shufflevector(
          *reinterpret_cast<const v8bf*>(&Vt[db * 16 + ln][half * 8]),
          *reinterpret_cast<const v8bf*>(&Vt[db * 16 + ln][16 + half * 8]),
          0, 1, 2, 3, 4, 5, 6, 7, 8, 9, 10, 11, 12, 13, 14, 15);
#pragma unroll
      for (int s = 0; s < QSTRIPE; ++s)
        acc[s][db] = __builtin_amdgcn_wmma_f32_16x16x32_bf16(
            false, pa[s], false, bv, (short)0, acc[s][db], false, false);
    }
  }

  // ---- epilogue: normalize by row sums, store f32 output ----
#pragma unroll
  for (int s = 0; s < QSTRIPE; ++s) {
#pragma unroll
    for (int r = 0; r < 8; ++r) {
      const float inv = 1.0f / l_i[s][r];
      const int   row = qrow0 + s * 16 + r + 8 * half;
      float* orow = ob + (size_t)row * ATTN_D;
#pragma unroll
      for (int db = 0; db < 4; ++db)
        orow[db * 16 + ln] = acc[s][db][r] * inv;
    }
  }
}

extern "C" void kernel_launch(void* const* d_in, const int* in_sizes, int n_in,
                              void* d_out, int out_size, void* d_ws, size_t ws_size,
                              hipStream_t stream) {
  (void)in_sizes; (void)n_in; (void)out_size; (void)d_ws; (void)ws_size;
  const float* q = (const float*)d_in[0];
  const float* k = (const float*)d_in[1];
  const float* v = (const float*)d_in[2];
  float*       o = (float*)d_out;
  dim3 grid(ATTN_S / Q_TILE, ATTN_B * ATTN_H);
  dim3 block(128);
  hipLaunchKernelGGL(flash_attn_fwd_gfx1250, grid, block, 0, stream, q, k, v, o);
}
